// RWKV7Block_47485158424860
// MI455X (gfx1250) — compile-verified, matrix operand order checked by reading, not run on hardware
//
#include <hip/hip_runtime.h>
#include <hip/hip_bf16.h>

typedef __attribute__((ext_vector_type(16))) _Float16 v16h;
typedef __attribute__((ext_vector_type(8)))  float    v8f;

union FragU { v16h h; unsigned u[8]; };

#define BK 64
#define TS 72   // halves per LDS row: 144B rows (16B aligned), 36-dword stride -> conflict-free frag reads

#if __has_builtin(__builtin_amdgcn_global_load_async_to_lds_b128)
#define HAVE_ASYNC_LDS 1
typedef int v4i_t __attribute__((vector_size(16)));
typedef __attribute__((address_space(1))) v4i_t* as1_v4i;
typedef __attribute__((address_space(3))) v4i_t* as3_v4i;
#define AS1(p) ((as1_v4i)(unsigned long long)(uintptr_t)(p))
#define AS3(p) ((as3_v4i)(unsigned)(uintptr_t)(p))
#else
#define HAVE_ASYNC_LDS 0
#endif

__device__ __forceinline__ void wait_async0() {
#if __has_builtin(__builtin_amdgcn_s_wait_asynccnt)
  __builtin_amdgcn_s_wait_asynccnt(0);
#else
  asm volatile("s_wait_asynccnt 0" ::: "memory");
#endif
}

__device__ __forceinline__ float sigmf(float x) { return 1.0f / (1.0f + __expf(-x)); }

// ---------------------------------------------------------------------------
// Epilogue modes:
// 0: f32 plain            1: f16 plain          2: f16 tanh       3: f16 sigmoid
// 4: f32 sigmoid(v+bias)  5: f32 sigmoid(v+bias)*exp(-0.5)
// 6: f32 v+resid          7: f32 v+bias+resid
// ---------------------------------------------------------------------------
template<int EPI>
__device__ __forceinline__ void epi_store(float* Cf, _Float16* Ch,
                                          const float* bias, const float* resid,
                                          int row, int col, int N, float v) {
  size_t idx = (size_t)row * N + col;
  if constexpr (EPI == 0) Cf[idx] = v;
  else if constexpr (EPI == 1) Ch[idx] = (_Float16)v;
  else if constexpr (EPI == 2) Ch[idx] = (_Float16)tanhf(v);
  else if constexpr (EPI == 3) Ch[idx] = (_Float16)sigmf(v);
  else if constexpr (EPI == 4) Cf[idx] = sigmf(v + bias[col]);
  else if constexpr (EPI == 5) Cf[idx] = 0.60653065971263342f * sigmf(v + bias[col]);
  else if constexpr (EPI == 6) Cf[idx] = v + resid[idx];
  else if constexpr (EPI == 7) Cf[idx] = v + bias[col] + resid[idx];
}

// C[M,N] = A[M,K] (f16, row-major) * Bt[N,K]^T (f16, N-major), f32 WMMA accum.
// 256 threads (8 waves), 64x64 tile, BK=64, double-buffered LDS, async-to-LDS
// when available. Wave w: M sub-tile 16*(w&3), N pair 32*(w>>2)+{0,16}.
template<int EPI>
__global__ __launch_bounds__(256) void gemm_wmma_kernel(
    const _Float16* __restrict__ A, const _Float16* __restrict__ Bt,
    float* __restrict__ Cf, _Float16* __restrict__ Ch,
    const float* __restrict__ bias, const float* __restrict__ resid,
    int M, int N, int K)
{
  __shared__ __align__(16) _Float16 As[2][64 * TS];
  __shared__ __align__(16) _Float16 Bs[2][64 * TS];

  const int tid  = threadIdx.x;
  const int lane = tid & 31;
  const int wave = tid >> 5;
  const int m0 = blockIdx.y * 64;
  const int n0 = blockIdx.x * 64;
  const int mw = 16 * (wave & 3);
  const int nw = 32 * (wave >> 2);
  const int mn   = lane & 15;
  const int hsel = lane >> 4;

  v8f acc0 = {}; v8f acc1 = {};

  const int lrow = tid >> 2;           // 0..63
  const int loff = (tid & 3) * 16;     // halves; two uint4 chunks at loff, loff+8

  const _Float16* Ag = A  + (size_t)(m0 + lrow) * K + loff;
  const _Float16* Bg = Bt + (size_t)(n0 + lrow) * K + loff;
  const int nk = K / BK;

#if !HAVE_ASYNC_LDS
  uint4 ra0, ra1, rb0, rb1;
#endif

  auto compute = [&](int buf) {
    const unsigned* arp  = (const unsigned*)&As[buf][(mw + mn) * TS];
    const unsigned* brp0 = (const unsigned*)&Bs[buf][(nw + mn) * TS];
    const unsigned* brp1 = (const unsigned*)&Bs[buf][(nw + 16 + mn) * TS];
    #pragma unroll
    for (int s = 0; s < 2; ++s) {
      const int db = 16 * s + hsel * 4;
      FragU af, bf0, bf1;
      #pragma unroll
      for (int i = 0; i < 4; ++i) {
        af.u[i]  = arp [db + i];  af.u[4+i]  = arp [db + 8 + i];
        bf0.u[i] = brp0[db + i];  bf0.u[4+i] = brp0[db + 8 + i];
        bf1.u[i] = brp1[db + i];  bf1.u[4+i] = brp1[db + 8 + i];
      }
      acc0 = __builtin_amdgcn_wmma_f32_16x16x32_f16(false, af.h, false, bf0.h,
                                                    (short)0, acc0, false, false);
      acc1 = __builtin_amdgcn_wmma_f32_16x16x32_f16(false, af.h, false, bf1.h,
                                                    (short)0, acc1, false, false);
    }
  };

#if HAVE_ASYNC_LDS
  auto async_tile = [&](int kt, int buf) {
    const _Float16* ap = Ag + kt * BK;
    const _Float16* bp = Bg + kt * BK;
    _Float16* la = &As[buf][lrow * TS + loff];
    _Float16* lb = &Bs[buf][lrow * TS + loff];
    __builtin_amdgcn_global_load_async_to_lds_b128(AS1(ap),     AS3(la),     0, 0);
    __builtin_amdgcn_global_load_async_to_lds_b128(AS1(ap + 8), AS3(la + 8), 0, 0);
    __builtin_amdgcn_global_load_async_to_lds_b128(AS1(bp),     AS3(lb),     0, 0);
    __builtin_amdgcn_global_load_async_to_lds_b128(AS1(bp + 8), AS3(lb + 8), 0, 0);
  };

  async_tile(0, 0);
  int cur = 0;
  for (int kt = 0; kt < nk; ++kt) {
    wait_async0();                      // our buf[cur] writes have landed in LDS
    __syncthreads();                    // everyone's have
    if (kt + 1 < nk) async_tile(kt + 1, cur ^ 1);
    compute(cur);
    cur ^= 1;
  }
#else
  auto gload = [&](int kt) {
    const _Float16* ap = Ag + kt * BK;
    const _Float16* bp = Bg + kt * BK;
    ra0 = *(const uint4*)ap;  ra1 = *(const uint4*)(ap + 8);
    rb0 = *(const uint4*)bp;  rb1 = *(const uint4*)(bp + 8);
  };
  auto sstore = [&](int buf) {
    *(uint4*)&As[buf][lrow * TS + loff]     = ra0;
    *(uint4*)&As[buf][lrow * TS + loff + 8] = ra1;
    *(uint4*)&Bs[buf][lrow * TS + loff]     = rb0;
    *(uint4*)&Bs[buf][lrow * TS + loff + 8] = rb1;
  };
  gload(0); sstore(0);
  int cur = 0;
  for (int kt = 0; kt < nk; ++kt) {
    __syncthreads();
    if (kt + 1 < nk) gload(kt + 1);     // issue next-tile global loads early
    compute(cur);
    if (kt + 1 < nk) sstore(cur ^ 1);
    cur ^= 1;
  }
#endif

  const int crow = m0 + mw + hsel * 8;
  const int col0 = n0 + nw + mn;
  #pragma unroll
  for (int j = 0; j < 8; ++j) {
    epi_store<EPI>(Cf, Ch, bias, resid, crow + j, col0,      N, acc0[j]);
    epi_store<EPI>(Cf, Ch, bias, resid, crow + j, col0 + 16, N, acc1[j]);
  }
}

// ---------------------------------------------------------------------------
// Weight convert+transpose: W[K,N] f32 -> Wt[N,K] f16
// ---------------------------------------------------------------------------
__global__ void transpose_kernel(const float* __restrict__ W, _Float16* __restrict__ Wt,
                                 int K, int N) {
  int idx = blockIdx.x * blockDim.x + threadIdx.x;
  if (idx >= K * N) return;
  int k = idx / N, n = idx % N;
  Wt[(size_t)n * K + k] = (_Float16)W[idx];
}

// ---------------------------------------------------------------------------
// LayerNorm over D=1024. One block per row, 4 elems/thread.
// ---------------------------------------------------------------------------
template<typename TO>
__global__ __launch_bounds__(256) void ln_kernel(const float* __restrict__ x,
    const float* __restrict__ w, const float* __restrict__ b, TO* __restrict__ y) {
  const int row = blockIdx.x;
  const int tid = threadIdx.x;
  const float* xr = x + (size_t)row * 1024;
  float v0[4], s = 0.f, s2 = 0.f;
  #pragma unroll
  for (int i = 0; i < 4; ++i) { float t = xr[tid * 4 + i]; v0[i] = t; s += t; s2 += t * t; }
  __shared__ float red[256], red2[256];
  red[tid] = s; red2[tid] = s2;
  __syncthreads();
  for (int off = 128; off > 0; off >>= 1) {
    if (tid < off) { red[tid] += red[tid + off]; red2[tid] += red2[tid + off]; }
    __syncthreads();
  }
  float mu  = red[0] * (1.0f / 1024.0f);
  float var = red2[0] * (1.0f / 1024.0f) - mu * mu;
  float inv = rsqrtf(var + 1e-5f);
  #pragma unroll
  for (int i = 0; i < 4; ++i) {
    int c = tid * 4 + i;
    y[(size_t)row * 1024 + c] = (TO)((v0[i] - mu) * inv * w[c] + b[c]);
  }
}

// ---------------------------------------------------------------------------
// Token-shift mixes: out = xn + (xn[t-1]-xn[t]) * mix, six streams, f16 out.
// ---------------------------------------------------------------------------
__global__ __launch_bounds__(256) void mix_kernel(const float* __restrict__ xn,
    const float* __restrict__ mr, const float* __restrict__ mw, const float* __restrict__ mk,
    const float* __restrict__ mv, const float* __restrict__ ma, const float* __restrict__ mg,
    _Float16* __restrict__ oxr, _Float16* __restrict__ oxw, _Float16* __restrict__ oxk,
    _Float16* __restrict__ oxv, _Float16* __restrict__ oxa, _Float16* __restrict__ oxg,
    int T) {
  const int row = blockIdx.x;
  const int t = row % T;
  const int tid = threadIdx.x;
  #pragma unroll
  for (int i = 0; i < 4; ++i) {
    int c = tid * 4 + i;
    size_t idx = (size_t)row * 1024 + c;
    float cur = xn[idx];
    float prev = (t > 0) ? xn[idx - 1024] : 0.0f;
    float d = prev - cur;
    oxr[idx] = (_Float16)(cur + d * mr[c]);
    oxw[idx] = (_Float16)(cur + d * mw[c]);
    oxk[idx] = (_Float16)(cur + d * mk[c]);
    oxv[idx] = (_Float16)(cur + d * mv[c]);
    oxa[idx] = (_Float16)(cur + d * ma[c]);
    oxg[idx] = (_Float16)(cur + d * mg[c]);
  }
}

// ---------------------------------------------------------------------------
// kk = normalize_per_head(k * k_k);  k <- k * (1 + (a-1)*k_a)   (in place)
// ---------------------------------------------------------------------------
__global__ __launch_bounds__(256) void kkprep_kernel(float* __restrict__ k,
    const float* __restrict__ a, const float* __restrict__ k_k,
    const float* __restrict__ k_a, float* __restrict__ kk) {
  const int row = blockIdx.x;
  const int tid = threadIdx.x;
  const int c0 = tid * 4;
  float kv[4], kke[4], ss = 0.f;
  #pragma unroll
  for (int i = 0; i < 4; ++i) {
    int c = c0 + i; size_t idx = (size_t)row * 1024 + c;
    kv[i] = k[idx]; kke[i] = kv[i] * k_k[c]; ss += kke[i] * kke[i];
  }
  __shared__ float part[256];
  __shared__ float hnorm[16];
  part[tid] = ss;
  __syncthreads();
  if (tid < 16) {
    float s = 0.f;
    #pragma unroll
    for (int j = 0; j < 16; ++j) s += part[tid * 16 + j];
    hnorm[tid] = fmaxf(sqrtf(s), 1e-12f);
  }
  __syncthreads();
  const float inv = 1.0f / hnorm[c0 >> 6];
  #pragma unroll
  for (int i = 0; i < 4; ++i) {
    int c = c0 + i; size_t idx = (size_t)row * 1024 + c;
    float av = a[idx];
    kk[idx] = kke[i] * inv;
    k[idx]  = kv[i] * (1.0f + (av - 1.0f) * k_a[c]);
  }
}

// ---------------------------------------------------------------------------
// Sequential WKV-7 scan. One block per (b,h). 256 threads; thread owns
// S[vr][kq..kq+15] in registers. Double-buffered LDS staging with register
// prefetch of timestep t+1 -> one barrier per step, load latency hidden.
// ---------------------------------------------------------------------------
__global__ __launch_bounds__(256) void scan_kernel(
    const float* __restrict__ r, const float* __restrict__ wd,
    const float* __restrict__ k, const float* __restrict__ v,
    const float* __restrict__ kk, const float* __restrict__ a,
    float* __restrict__ o, int T, int H) {
  const int b = blockIdx.x / H, h = blockIdx.x % H;
  const int tid = threadIdx.x;
  const int vr = tid >> 2;
  const int kq = (tid & 3) * 16;
  float S[16];
  #pragma unroll
  for (int i = 0; i < 16; ++i) S[i] = 0.0f;

  // layout: [0:64)=r [64:128)=wd [128:192)=k [192:256)=v [256:320)=kk [320:384)=a
  __shared__ float buf[2][384];
  const int g = tid >> 6, e = tid & 63;
  const float* s0 = r;  int o0 = 0;
  const float* s1 = wd; int o1 = 64;
  if (g == 1) { s0 = k;  o0 = 128; s1 = v; o1 = 192; }
  else if (g == 2) { s0 = kk; o0 = 256; s1 = a; o1 = 320; }
  const bool loader = (g < 3);
  const size_t hb = (size_t)b * T * 1024 + h * 64 + e;

  float p0 = 0.f, p1 = 0.f;
  if (loader) { p0 = s0[hb]; p1 = s1[hb]; buf[0][o0 + e] = p0; buf[0][o1 + e] = p1; }
  int cur = 0;

  for (int t = 0; t < T; ++t) {
    __syncthreads();                               // buf[cur] ready for all
    if (loader && t + 1 < T) {                     // issue t+1 global loads early
      size_t nb = hb + (size_t)(t + 1) * 1024;
      p0 = s0[nb]; p1 = s1[nb];
    }

    const float* rs  = &buf[cur][0];
    const float* wds = &buf[cur][64];
    const float* ks  = &buf[cur][128];
    const float* vs  = &buf[cur][192];
    const float* kks = &buf[cur][256];
    const float* as_ = &buf[cur][320];

    float sab = 0.0f;
    #pragma unroll
    for (int i = 0; i < 16; ++i) sab += S[i] * kks[kq + i];
    sab += __shfl_xor(sab, 1);
    sab += __shfl_xor(sab, 2);

    const float vt = vs[vr];
    float op = 0.0f;
    #pragma unroll
    for (int i = 0; i < 16; ++i) {
      const int c = kq + i;
      S[i] = S[i] * wds[c] - sab * (kks[c] * as_[c]) + vt * ks[c];
      op += S[i] * rs[c];
    }
    op += __shfl_xor(op, 1);
    op += __shfl_xor(op, 2);
    if ((tid & 3) == 0) o[(size_t)(b * T + t) * 1024 + h * 64 + vr] = op;

    if (loader && t + 1 < T) { buf[cur ^ 1][o0 + e] = p0; buf[cur ^ 1][o1 + e] = p1; }
    cur ^= 1;
  }
}

// ---------------------------------------------------------------------------
// Per-head group norm + bonus + gate -> f16 Wo input.
// ---------------------------------------------------------------------------
__global__ __launch_bounds__(256) void outnorm_kernel(
    const float* __restrict__ o, const float* __restrict__ r, const float* __restrict__ k,
    const float* __restrict__ v, const float* __restrict__ g,
    const float* __restrict__ r_k, const float* __restrict__ gn_w,
    const float* __restrict__ gn_b, _Float16* __restrict__ gated) {
  const int row = blockIdx.x;
  const int tid = threadIdx.x;
  const int c0 = tid * 4;
  float ov[4], s = 0.f, s2 = 0.f, bp = 0.f;
  #pragma unroll
  for (int i = 0; i < 4; ++i) {
    int c = c0 + i; size_t idx = (size_t)row * 1024 + c;
    float t = o[idx]; ov[i] = t; s += t; s2 += t * t;
    bp += r[idx] * k[idx] * r_k[c];
  }
  __shared__ float ps[256], ps2[256], pb[256];
  __shared__ float hmu[16], hinv[16], hb[16];
  ps[tid] = s; ps2[tid] = s2; pb[tid] = bp;
  __syncthreads();
  if (tid < 16) {
    float S = 0.f, S2 = 0.f, Bn = 0.f;
    #pragma unroll
    for (int j = 0; j < 16; ++j) { S += ps[tid*16+j]; S2 += ps2[tid*16+j]; Bn += pb[tid*16+j]; }
    float mu = S * (1.0f / 64.0f);
    float var = S2 * (1.0f / 64.0f) - mu * mu;
    hmu[tid] = mu; hinv[tid] = rsqrtf(var + 64e-5f); hb[tid] = Bn;
  }
  __syncthreads();
  const int h = c0 >> 6;
  #pragma unroll
  for (int i = 0; i < 4; ++i) {
    int c = c0 + i; size_t idx = (size_t)row * 1024 + c;
    float oo = (ov[i] - hmu[h]) * hinv[h] * gn_w[c] + gn_b[c];
    oo += hb[h] * v[idx];
    gated[idx] = (_Float16)(oo * g[idx]);
  }
}

// ---------------------------------------------------------------------------
extern "C" void kernel_launch(void* const* d_in, const int* in_sizes, int n_in,
                              void* d_out, int out_size, void* d_ws, size_t ws_size,
                              hipStream_t stream) {
  (void)in_sizes; (void)n_in; (void)out_size; (void)ws_size;
  const int Bn = 4, T = 1024, D = 1024, H = 16;
  const int M = Bn * T;

  const float* x     = (const float*)d_in[0];
  const float* ln1_w = (const float*)d_in[1];
  const float* ln1_b = (const float*)d_in[2];
  const float* ln2_w = (const float*)d_in[3];
  const float* ln2_b = (const float*)d_in[4];
  const float* x_r = (const float*)d_in[5];
  const float* x_w = (const float*)d_in[6];
  const float* x_k = (const float*)d_in[7];
  const float* x_v = (const float*)d_in[8];
  const float* x_a = (const float*)d_in[9];
  const float* x_g = (const float*)d_in[10];
  const float* Wr = (const float*)d_in[11];
  const float* Wk = (const float*)d_in[12];
  const float* Wv = (const float*)d_in[13];
  const float* w0 = (const float*)d_in[14];
  const float* w1 = (const float*)d_in[15];
  const float* w2 = (const float*)d_in[16];
  const float* a0 = (const float*)d_in[17];
  const float* a1 = (const float*)d_in[18];
  const float* a2 = (const float*)d_in[19];
  const float* g1 = (const float*)d_in[20];
  const float* g2 = (const float*)d_in[21];
  const float* k_k = (const float*)d_in[22];
  const float* k_a = (const float*)d_in[23];
  const float* r_k = (const float*)d_in[24];
  const float* gn_w = (const float*)d_in[25];
  const float* gn_b = (const float*)d_in[26];
  const float* Wo = (const float*)d_in[27];
  const float* mlp_w = (const float*)d_in[28];
  const float* mlp_b = (const float*)d_in[29];

  // ---- workspace carve -----------------------------------------------------
  char* p = (char*)d_ws;
  auto alloc = [&](size_t bytes) { void* q = p; p += (bytes + 255) & ~(size_t)255; return q; };
  const size_t MD = (size_t)M * D;

  float* xn  = (float*)alloc(MD * 4);
  float* rb  = (float*)alloc(MD * 4);
  float* kb  = (float*)alloc(MD * 4);
  float* wdb = (float*)alloc(MD * 4);
  float* ab  = (float*)alloc(MD * 4);
  float* gb  = (float*)alloc(MD * 4);
  float* kkb = (float*)alloc(MD * 4);
  float* ob  = (float*)alloc(MD * 4);
  float* x1b = (float*)alloc(MD * 4);

  _Float16* xr16 = (_Float16*)alloc(MD * 2);
  _Float16* xw16 = (_Float16*)alloc(MD * 2);
  _Float16* xk16 = (_Float16*)alloc(MD * 2);
  _Float16* xv16 = (_Float16*)alloc(MD * 2);
  _Float16* xa16 = (_Float16*)alloc(MD * 2);
  _Float16* xg16 = (_Float16*)alloc(MD * 2);

  _Float16* WrT  = (_Float16*)alloc((size_t)D * D * 2);
  _Float16* WkT  = (_Float16*)alloc((size_t)D * D * 2);
  _Float16* WvT  = (_Float16*)alloc((size_t)D * D * 2);
  _Float16* WoT  = (_Float16*)alloc((size_t)D * D * 2);
  _Float16* MlpT = (_Float16*)alloc((size_t)D * D * 2);
  _Float16* w1T  = (_Float16*)alloc((size_t)64  * D * 2);
  _Float16* w2T  = (_Float16*)alloc((size_t)D * 64  * 2);
  _Float16* a1T  = (_Float16*)alloc((size_t)64  * D * 2);
  _Float16* a2T  = (_Float16*)alloc((size_t)D * 64  * 2);
  _Float16* g1T  = (_Float16*)alloc((size_t)128 * D * 2);
  _Float16* g2T  = (_Float16*)alloc((size_t)D * 128 * 2);

  _Float16* wtmp = (_Float16*)alloc((size_t)M * 64  * 2);
  _Float16* atmp = (_Float16*)alloc((size_t)M * 64  * 2);
  _Float16* gtmp = (_Float16*)alloc((size_t)M * 128 * 2);

  float*    vb    = (float*)d_out + MD;   // second output: v_first (written by v-GEMM)
  float*    xout  = (float*)d_out;
  _Float16* gated = xr16;                 // reuse
  _Float16* ln2x  = xk16;                 // reuse

  // ---- weight convert/transpose -------------------------------------------
  auto xpose = [&](const float* W, _Float16* Wt, int K, int N) {
    int n = K * N;
    transpose_kernel<<<(n + 255) / 256, 256, 0, stream>>>(W, Wt, K, N);
  };
  xpose(Wr, WrT, D, D);   xpose(Wk, WkT, D, D);   xpose(Wv, WvT, D, D);
  xpose(Wo, WoT, D, D);   xpose(mlp_w, MlpT, D, D);
  xpose(w1, w1T, D, 64);  xpose(w2, w2T, 64, D);
  xpose(a1, a1T, D, 64);  xpose(a2, a2T, 64, D);
  xpose(g1, g1T, D, 128); xpose(g2, g2T, 128, D);

  // ---- LN1 + token-shift mixes --------------------------------------------
  ln_kernel<float><<<M, 256, 0, stream>>>(x, ln1_w, ln1_b, xn);
  mix_kernel<<<M, 256, 0, stream>>>(xn, x_r, x_w, x_k, x_v, x_a, x_g,
                                    xr16, xw16, xk16, xv16, xa16, xg16, T);

  // ---- projections (WMMA) --------------------------------------------------
  dim3 blk(256);
  auto grid = [&](int N_) { return dim3(N_ / 64, M / 64); };

  gemm_wmma_kernel<0><<<grid(D), blk, 0, stream>>>(xr16, WrT, rb, nullptr, nullptr, nullptr, M, D, D);
  gemm_wmma_kernel<0><<<grid(D), blk, 0, stream>>>(xk16, WkT, kb, nullptr, nullptr, nullptr, M, D, D);
  gemm_wmma_kernel<0><<<grid(D), blk, 0, stream>>>(xv16, WvT, vb, nullptr, nullptr, nullptr, M, D, D);

  // decay LoRA: wd = sigmoid(w0 + tanh(xw@w1)@w2) * exp(-0.5)   [softplus folded]
  gemm_wmma_kernel<2><<<grid(64), blk, 0, stream>>>(xw16, w1T, nullptr, wtmp, nullptr, nullptr, M, 64, D);
  gemm_wmma_kernel<5><<<grid(D),  blk, 0, stream>>>(wtmp, w2T, wdb, nullptr, w0, nullptr, M, D, 64);

  // a LoRA: a = sigmoid(a0 + (xa@a1)@a2)
  gemm_wmma_kernel<1><<<grid(64), blk, 0, stream>>>(xa16, a1T, nullptr, atmp, nullptr, nullptr, M, 64, D);
  gemm_wmma_kernel<4><<<grid(D),  blk, 0, stream>>>(atmp, a2T, ab, nullptr, a0, nullptr, M, D, 64);

  // gate: g = sigmoid(xg@g1) @ g2
  gemm_wmma_kernel<3><<<grid(128), blk, 0, stream>>>(xg16, g1T, nullptr, gtmp, nullptr, nullptr, M, 128, D);
  gemm_wmma_kernel<0><<<grid(D),   blk, 0, stream>>>(gtmp, g2T, gb, nullptr, nullptr, nullptr, M, D, 128);

  // ---- kk normalize + k update --------------------------------------------
  kkprep_kernel<<<M, 256, 0, stream>>>(kb, ab, k_k, k_a, kkb);

  // ---- sequential scan -----------------------------------------------------
  scan_kernel<<<Bn * H, 256, 0, stream>>>(rb, wdb, kb, vb, kkb, ab, ob, T, H);

  // ---- group norm + bonus + gate ------------------------------------------
  outnorm_kernel<<<M, 256, 0, stream>>>(ob, rb, kb, vb, gb, r_k, gn_w, gn_b, gated);

  // ---- output projection + residual ---------------------------------------
  gemm_wmma_kernel<6><<<grid(D), blk, 0, stream>>>(gated, WoT, x1b, nullptr, nullptr, x, M, D, D);

  // ---- channel mix: x1 + (ln2(x1) @ mlp_w + mlp_b) -------------------------
  ln_kernel<_Float16><<<M, 256, 0, stream>>>(x1b, ln2_w, ln2_b, ln2x);
  gemm_wmma_kernel<7><<<grid(D), blk, 0, stream>>>(ln2x, MlpT, xout, nullptr, mlp_b, x1b, M, D, D);
}
